// TopKSAE_71219147702781
// MI455X (gfx1250) — matrix operand order repaired and checked by reading, not compiled, not run on hardware
//
#include <hip/hip_runtime.h>

// ---------------- problem constants ----------------
#define D_MODEL 1024
#define D_SAE   16384
#define NROWS   8192
#define KTOP    64

typedef __bf16 bf16;
typedef __attribute__((ext_vector_type(16))) bf16  v16bf;
typedef __attribute__((ext_vector_type(8)))  bf16  v8bf;
typedef __attribute__((ext_vector_type(8)))  float v8f;

// round-to-nearest-even float -> bf16 via bit ops
__device__ inline bf16 f2bf(float f) {
  unsigned u = __float_as_uint(f);
  unsigned r = (u + 0x7FFFu + ((u >> 16) & 1u)) >> 16;
  unsigned short s = (unsigned short)r;
  bf16 b;
  __builtin_memcpy(&b, &s, 2);
  return b;
}

__device__ inline void store4_bf16(bf16* dst, float a, float b, float c, float d) {
  union { bf16 h[4]; unsigned long long u; } p;
  p.h[0] = f2bf(a); p.h[1] = f2bf(b); p.h[2] = f2bf(c); p.h[3] = f2bf(d);
  *(unsigned long long*)dst = p.u;   // 8B aligned
}

// =====================================================================
// Prep kernels: one-time fp32 -> bf16 conversion into workspace.
// =====================================================================
__global__ __launch_bounds__(256) void prep_x(
    const float* __restrict__ x, const float* __restrict__ b_pre,
    bf16* __restrict__ xc)
{
  const size_t total4 = (size_t)NROWS * D_MODEL / 4;
  for (size_t i = (size_t)blockIdx.x * 256 + threadIdx.x; i < total4;
       i += (size_t)gridDim.x * 256) {
    const float4 a  = ((const float4*)x)[i];
    const int d     = (int)((i * 4) & (D_MODEL - 1));
    const float4 bp = *(const float4*)&b_pre[d];
    store4_bf16(&xc[i * 4], a.x - bp.x, a.y - bp.y, a.z - bp.z, a.w - bp.w);
  }
}

__global__ __launch_bounds__(256) void prep_w(
    const float* __restrict__ w, bf16* __restrict__ wc)
{
  const size_t total4 = (size_t)D_SAE * D_MODEL / 4;
  for (size_t i = (size_t)blockIdx.x * 256 + threadIdx.x; i < total4;
       i += (size_t)gridDim.x * 256) {
    const float4 a = ((const float4*)w)[i];
    store4_bf16(&wc[i * 4], a.x, a.y, a.z, a.w);
  }
}

// =====================================================================
// FAST encoder GEMM (prepped bf16 inputs):
//   block tile 64(M) x 256(S), K-step 32, 8 waves, wave tile 32x64
//   => 8 back-to-back v_wmma_f32_16x16x32_bf16 per wave per K-step.
// Staging: GLOBAL_LOAD_ASYNC_TO_LDS_B128 (ASYNCcnt DMA), 3-stage LDS
// pipeline: two stages in flight, s_wait_asynccnt 0x5 keeps one stage
// outstanding while the oldest is guaranteed landed (5 loads/stage/wave).
// =====================================================================
#define FBM 64
#define FBS 256
#define FBK 32

__global__ __launch_bounds__(256) void enc_gemm_async(
    const bf16* __restrict__ Xc, const bf16* __restrict__ We,
    const float* __restrict__ b_enc, float* __restrict__ Z)
{
  __shared__ __align__(32) bf16 sA[3][FBM][FBK];   // 3 x 4 KB
  __shared__ __align__(32) bf16 sW[3][FBS][FBK];   // 3 x 16 KB

  const int tid  = threadIdx.x;
  const int lane = tid & 31;
  const int wave = tid >> 5;       // 0..7
  const int half = lane >> 4;      // 0/1
  const int l16  = lane & 15;
  const int sw   = wave & 3;       // S sub-block (64 cols each)
  const int mw   = wave >> 2;      // M sub-block (32 rows each)

  const int row0 = blockIdx.y * FBM;
  const int s0   = blockIdx.x * FBS;

  // per-thread staging chunks (16B of bf16 each): A 1 chunk, W 4 chunks
  const int ar = tid >> 2, ac = (tid & 3) * 8;
  const int wrb = tid >> 2, wc = (tid & 3) * 8;

  auto stage = [&](int k0, int buf) {
    {
      unsigned lo = (unsigned)(size_t)&sA[buf][ar][ac];
      unsigned long long ga =
          (unsigned long long)(size_t)(Xc + (size_t)(row0 + ar) * D_MODEL + k0 + ac);
      asm volatile("global_load_async_to_lds_b128 %0, %1, off"
                   :: "v"(lo), "v"(ga) : "memory");
    }
    #pragma unroll
    for (int q = 0; q < 4; q++) {
      const int r = wrb + q * 64;
      unsigned lo = (unsigned)(size_t)&sW[buf][r][wc];
      unsigned long long ga =
          (unsigned long long)(size_t)(We + (size_t)(s0 + r) * D_MODEL + k0 + wc);
      asm volatile("global_load_async_to_lds_b128 %0, %1, off"
                   :: "v"(lo), "v"(ga) : "memory");
    }
  };

  v8f acc[2][4] = {};

  stage(0, 0);
  stage(FBK, 1);

  const int NT = D_MODEL / FBK;   // 32
  int cur = 0;
  for (int kt = 0; kt < NT; kt++) {
    // oldest in-flight stage (buffer `cur`) must be resident:
    //   steady state: 2 stages in flight (10 loads) -> wait <= 5
    //   last iter:    1 stage in flight  (5 loads)  -> wait <= 0
    if (kt == NT - 1)
      asm volatile("s_wait_asynccnt 0x0" ::: "memory");
    else
      asm volatile("s_wait_asynccnt 0x5" ::: "memory");
    __syncthreads();

    // refill buffer (cur+2)%3 == buffer used at kt-1, free after barrier
    if (kt + 2 < NT) {
      int nb = cur + 2; if (nb >= 3) nb -= 3;
      stage((kt + 2) * FBK, nb);
    }

    // ---- preload ALL fragments, then 8 WMMAs back-to-back ----
    // A layout: lane m=l16, k = 16*half..16*half+15 (contiguous 32B)
    v16bf af[2];
    #pragma unroll
    for (int mi = 0; mi < 2; mi++) {
      const int m = mw * 32 + mi * 16 + l16;
      v8bf alo = *(const v8bf*)&sA[cur][m][16 * half];
      v8bf ahi = *(const v8bf*)&sA[cur][m][16 * half + 8];
      af[mi] = __builtin_shufflevector(alo, ahi,
          0, 1, 2, 3, 4, 5, 6, 7, 8, 9, 10, 11, 12, 13, 14, 15);
    }
    // B layout: lane n=l16, k = 16*half..16*half+15 (contiguous 32B)
    v16bf bfr[4];
    #pragma unroll
    for (int t = 0; t < 4; t++) {
      const int sn = sw * 64 + t * 16 + l16;
      bfr[t] = *(const v16bf*)&sW[cur][sn][16 * half];
    }
    #pragma unroll
    for (int t = 0; t < 4; t++)
      #pragma unroll
      for (int mi = 0; mi < 2; mi++)
        acc[mi][t] = __builtin_amdgcn_wmma_f32_16x16x32_bf16(
            false, af[mi], false, bfr[t], (short)0, acc[mi][t], false, false);

    cur++; if (cur >= 3) cur = 0;
  }

  // epilogue: D layout n=l16, vgpr r -> m = r + 8*half
  #pragma unroll
  for (int mi = 0; mi < 2; mi++) {
    #pragma unroll
    for (int t = 0; t < 4; t++) {
      const int sn = s0 + sw * 64 + t * 16 + l16;
      const float be = b_enc[sn];
      #pragma unroll
      for (int r = 0; r < 8; r++) {
        const int mrow = row0 + mw * 32 + mi * 16 + r + 8 * half;
        Z[(size_t)mrow * D_SAE + sn] = acc[mi][t][r] + be;
      }
    }
  }
}

// =====================================================================
// FALLBACK encoder GEMM (small workspace): inline fp32->bf16 conversion.
// =====================================================================
#define BM 64
#define BS 128
#define BK 32

__global__ __launch_bounds__(256) void enc_gemm_inline(
    const float* __restrict__ x, const float* __restrict__ b_pre,
    const float* __restrict__ W_enc, const float* __restrict__ b_enc,
    float* __restrict__ Z)
{
  __shared__ __align__(32) bf16 sA[BM][BK];
  __shared__ __align__(32) bf16 sW[BS][BK];

  const int tid   = threadIdx.x;
  const int lane  = tid & 31;
  const int wave  = tid >> 5;
  const int half  = lane >> 4;
  const int l16   = lane & 15;
  const int mwave = wave & 3;
  const int swave = wave >> 2;

  const int row0 = blockIdx.y * BM;
  const int s0   = blockIdx.x * BS;

  v8f acc[4] = {};

  for (int k0 = 0; k0 < D_MODEL; k0 += BK) {
    #pragma unroll
    for (int q = 0; q < 2; q++) {
      int c  = tid + q * 256;
      int r  = c >> 3;
      int cc = (c & 7) * 4;
      const float4 a  = *(const float4*)&x[(size_t)(row0 + r) * D_MODEL + k0 + cc];
      const float4 bp = *(const float4*)&b_pre[k0 + cc];
      store4_bf16(&sA[r][cc], a.x - bp.x, a.y - bp.y, a.z - bp.z, a.w - bp.w);
    }
    #pragma unroll
    for (int q = 0; q < 4; q++) {
      int c  = tid + q * 256;
      int r  = c >> 3;
      int cc = (c & 7) * 4;
      const float4 w = *(const float4*)&W_enc[(size_t)(s0 + r) * D_MODEL + k0 + cc];
      store4_bf16(&sW[r][cc], w.x, w.y, w.z, w.w);
      if (k0 + BK < D_MODEL)
        __builtin_prefetch(&W_enc[(size_t)(s0 + r) * D_MODEL + k0 + BK + cc], 0, 3);
    }
    __syncthreads();

    const int m = mwave * 16 + l16;
    v8bf alo = *(const v8bf*)&sA[m][16 * half];
    v8bf ahi = *(const v8bf*)&sA[m][16 * half + 8];
    v16bf afrag = __builtin_shufflevector(alo, ahi,
        0, 1, 2, 3, 4, 5, 6, 7, 8, 9, 10, 11, 12, 13, 14, 15);

    #pragma unroll
    for (int t = 0; t < 4; t++) {
      const int sn = swave * 64 + t * 16 + l16;
      v16bf bfr = *(const v16bf*)&sW[sn][16 * half];
      acc[t] = __builtin_amdgcn_wmma_f32_16x16x32_bf16(
          false, afrag, false, bfr, (short)0, acc[t], false, false);
    }
    __syncthreads();
  }

  #pragma unroll
  for (int t = 0; t < 4; t++) {
    const int sn = s0 + swave * 64 + t * 16 + l16;
    const float be = b_enc[sn];
    #pragma unroll
    for (int r = 0; r < 8; r++) {
      const int mrow = row0 + mwave * 16 + r + 8 * half;
      Z[(size_t)mrow * D_SAE + sn] = acc[t][r] + be;
    }
  }
}

// =====================================================================
// Per-row exact top-K via 4-pass radix select; sparsifies Z in place,
// emits compact (val, idx) lists. Ties accepted in ascending index order.
// =====================================================================
__device__ inline unsigned f2key(float f) {
  unsigned u = __float_as_uint(f);
  return (u & 0x80000000u) ? ~u : (u | 0x80000000u);
}

__global__ __launch_bounds__(256) void topk_kernel(
    float* __restrict__ Z, float* __restrict__ vals, int* __restrict__ idxs)
{
  const int row = blockIdx.x, tid = threadIdx.x;
  float* zr = Z + (size_t)row * D_SAE;

  __shared__ unsigned hist[256];
  __shared__ unsigned tiebase[256];
  __shared__ int s_bucket, s_above;
  __shared__ unsigned outCnt;

  if (tid < KTOP) {
    vals[(size_t)row * KTOP + tid] = 0.0f;
    idxs[(size_t)row * KTOP + tid] = 0;
  }

  unsigned prefix = 0;
  int remaining = KTOP;
  for (int shift = 24; shift >= 0; shift -= 8) {
    hist[tid] = 0;
    __syncthreads();
    const unsigned himask = (shift == 24) ? 0u : (0xFFFFFFFFu << (shift + 8));
    for (int i = tid; i < D_SAE; i += 256) {
      unsigned key = f2key(zr[i]);
      if ((key & himask) == prefix)
        atomicAdd(&hist[(key >> shift) & 0xFFu], 1u);
    }
    __syncthreads();
    if (tid == 0) {
      int cum = 0, b;
      for (b = 255; b >= 0; b--) {
        int h = (int)hist[b];
        if (cum + h >= remaining) break;
        cum += h;
      }
      if (b < 0) b = 0;
      s_bucket = b; s_above = cum;
    }
    __syncthreads();
    prefix |= ((unsigned)s_bucket) << shift;
    remaining -= s_above;
    __syncthreads();
  }
  const unsigned thresh = prefix;
  const int T = remaining;

  const int CH = D_SAE / 256;
  const int base = tid * CH;
  unsigned myT = 0;
  for (int i = 0; i < CH; i++)
    if (f2key(zr[base + i]) == thresh) myT++;
  tiebase[tid] = myT;
  if (tid == 0) outCnt = 0;
  __syncthreads();
  if (tid == 0) {
    unsigned s = 0;
    for (int t2 = 0; t2 < 256; t2++) { unsigned c = tiebase[t2]; tiebase[t2] = s; s += c; }
  }
  __syncthreads();

  unsigned tieRank = tiebase[tid];
  for (int i = 0; i < CH; i++) {
    const int gi = base + i;
    const float v = zr[gi];
    const unsigned key = f2key(v);
    bool keep;
    if (key > thresh)       keep = true;
    else if (key == thresh) { keep = (tieRank < (unsigned)T); tieRank++; }
    else                    keep = false;
    if (keep) {
      unsigned slot = atomicAdd(&outCnt, 1u);
      if (slot < KTOP) {
        vals[(size_t)row * KTOP + slot] = v;
        idxs[(size_t)row * KTOP + slot] = gi;
      }
    } else {
      zr[gi] = 0.0f;
    }
  }
}

// =====================================================================
// W_dec transpose: [D_MODEL][D_SAE] -> [D_SAE][D_MODEL]
// =====================================================================
__global__ __launch_bounds__(256) void wdec_transpose(
    const float* __restrict__ Wd, float* __restrict__ WdT)
{
  __shared__ float tile[32][33];
  const int tx = threadIdx.x & 31, ty = threadIdx.x >> 5;
  const int sbase = blockIdx.x * 32;
  const int dbase = blockIdx.y * 32;
  #pragma unroll
  for (int r = 0; r < 32; r += 8)
    tile[ty + r][tx] = Wd[(size_t)(dbase + ty + r) * D_SAE + sbase + tx];
  __syncthreads();
  #pragma unroll
  for (int r = 0; r < 32; r += 8)
    WdT[(size_t)(sbase + ty + r) * D_MODEL + dbase + tx] = tile[tx][ty + r];
}

// =====================================================================
// Sparse decode: recon[n,:] = sum_k vals[n,k] * W_dec[:, idx[n,k]] + b_dec
// =====================================================================
__global__ __launch_bounds__(256) void decode_kernel(
    const float* __restrict__ vals, const int* __restrict__ idxs,
    const float* __restrict__ Wd, const float* __restrict__ b_dec,
    float* __restrict__ recon, int transposed)
{
  const int row = blockIdx.x, tid = threadIdx.x;
  __shared__ float sv[KTOP];
  __shared__ int   si[KTOP];
  if (tid < KTOP) {
    sv[tid] = vals[(size_t)row * KTOP + tid];
    si[tid] = idxs[(size_t)row * KTOP + tid];
  }
  __syncthreads();

  const int d = tid * 4;
  float a0 = b_dec[d + 0], a1 = b_dec[d + 1], a2 = b_dec[d + 2], a3 = b_dec[d + 3];
  if (transposed) {
    #pragma unroll 4
    for (int k = 0; k < KTOP; k++) {
      const float v = sv[k];
      const float* w = &Wd[(size_t)si[k] * D_MODEL + d];
      a0 = fmaf(v, w[0], a0); a1 = fmaf(v, w[1], a1);
      a2 = fmaf(v, w[2], a2); a3 = fmaf(v, w[3], a3);
    }
  } else {
    for (int k = 0; k < KTOP; k++) {
      const float v = sv[k];
      const size_t s = (size_t)si[k];
      a0 = fmaf(v, Wd[(size_t)(d + 0) * D_SAE + s], a0);
      a1 = fmaf(v, Wd[(size_t)(d + 1) * D_SAE + s], a1);
      a2 = fmaf(v, Wd[(size_t)(d + 2) * D_SAE + s], a2);
      a3 = fmaf(v, Wd[(size_t)(d + 3) * D_SAE + s], a3);
    }
  }
  float* out = &recon[(size_t)row * D_MODEL + d];
  out[0] = a0; out[1] = a1; out[2] = a2; out[3] = a3;
}

// =====================================================================
extern "C" void kernel_launch(void* const* d_in, const int* in_sizes, int n_in,
                              void* d_out, int out_size, void* d_ws, size_t ws_size,
                              hipStream_t stream)
{
  const float* x     = (const float*)d_in[0];
  const float* b_pre = (const float*)d_in[1];
  const float* W_enc = (const float*)d_in[2];
  const float* b_enc = (const float*)d_in[3];
  const float* W_dec = (const float*)d_in[4];
  const float* b_dec = (const float*)d_in[5];

  float* recon = (float*)d_out;                       // [NROWS * D_MODEL]
  float* Z     = recon + (size_t)NROWS * D_MODEL;     // z_sparse region

  char* ws = (char*)d_ws;
  size_t off = 0;
  float* vals = (float*)(ws + off); off += (size_t)NROWS * KTOP * sizeof(float);
  int*   idxs = (int*)(ws + off);   off += (size_t)NROWS * KTOP * sizeof(int);
  bf16*  xc   = (bf16*)(ws + off);  off += (size_t)NROWS * D_MODEL * sizeof(bf16);
  bf16*  we   = (bf16*)(ws + off);  off += (size_t)D_SAE * D_MODEL * sizeof(bf16);
  const size_t needFast = off;
  float* WdT  = (float*)(ws + off);
  const size_t needT = off + (size_t)D_MODEL * D_SAE * sizeof(float);

  const int useFast = (ws_size >= needFast) ? 1 : 0;
  const int useT    = (ws_size >= needT) ? 1 : 0;

  if (useT)
    wdec_transpose<<<dim3(D_SAE / 32, D_MODEL / 32), 256, 0, stream>>>(W_dec, WdT);

  if (useFast) {
    prep_x<<<4096, 256, 0, stream>>>(x, b_pre, xc);
    prep_w<<<8192, 256, 0, stream>>>(W_enc, we);
    enc_gemm_async<<<dim3(D_SAE / FBS, NROWS / FBM), 256, 0, stream>>>(
        xc, we, b_enc, Z);
  } else {
    enc_gemm_inline<<<dim3(D_SAE / BS, NROWS / BM), 256, 0, stream>>>(
        x, b_pre, W_enc, b_enc, Z);
  }

  topk_kernel<<<NROWS, 256, 0, stream>>>(Z, vals, idxs);

  decode_kernel<<<NROWS, 256, 0, stream>>>(vals, idxs, useT ? WdT : W_dec,
                                           b_dec, recon, useT);
}
